// SelfAttention1D_52767968198890
// MI455X (gfx1250) — compile-verified, matrix operand order checked by reading, not run on hardware
//
#include <hip/hip_runtime.h>

// ---------------------------------------------------------------------------
// SelfAttention1D for MI455X (gfx1250): f16 WMMA everywhere, f32 accumulate.
// B=8, C=128, L=2048, H=4, D=32.  D==32 matches v_wmma_f32_16x16x32_f16 K.
// Attention uses 64-key blocks: 4 S-WMMAs + one softmax pass + 4 P.V-WMMAs,
// halving the per-key softmax VALU overhead vs a 32-key block.
// ---------------------------------------------------------------------------

typedef __attribute__((ext_vector_type(16))) _Float16 v16h;
typedef __attribute__((ext_vector_type(8)))  _Float16 v8h;
typedef __attribute__((ext_vector_type(8)))  float    v8f;

#define B_SZ   8
#define C_SZ   128
#define L_SEQ  2048
#define H_NUM  4
#define D_HD   32
#define QSCALE 0.17677669529663688f   // 32^-0.5

union FragU { v16h v; v8h h[2]; };

// Load a 16x32 fp16 fragment in CDNA5 WMMA A/B VGPR layout from row-major
// storage (ld = leading dim in elements).  Per ISA 7.12.2: lane (r = lane%16,
// khalf = lane/16) holds row r, K = [khalf*8, +8) in v0..3 and
// K = [16+khalf*8, +8) in v4..7  ->  two contiguous 16-byte loads per lane.
__device__ inline v16h load_frag(const _Float16* __restrict__ p, int ld,
                                 int row0, int k0) {
    int lane  = threadIdx.x & 31;
    int r     = lane & 15;
    int khalf = lane >> 4;
    const _Float16* base = p + (size_t)(row0 + r) * ld + k0 + khalf * 8;
    FragU f;
    f.h[0] = *(const v8h*)(base);
    f.h[1] = *(const v8h*)(base + 16);
    return f.v;
}

__device__ inline v8f wmma16(v16h a, v16h b, v8f c) {
    return __builtin_amdgcn_wmma_f32_16x16x32_f16(
        false, a, false, b, (short)0, c, false, false);
}

// ---------------------------------------------------------------------------
// Kernel 0a: x [B,C,L] f32  ->  xT [B,L,C] f16  (GEMM-B columns contiguous)
// ---------------------------------------------------------------------------
__global__ void k_cvt_x(const float* __restrict__ x, _Float16* __restrict__ xT) {
    size_t i = (size_t)blockIdx.x * blockDim.x + threadIdx.x;
    if (i >= (size_t)B_SZ * C_SZ * L_SEQ) return;
    int l = (int)(i % L_SEQ);
    int t = (int)(i / L_SEQ);
    int c = t % C_SZ;
    int b = t / C_SZ;
    xT[((size_t)b * L_SEQ + l) * C_SZ + c] = (_Float16)x[i];
}

// Kernel 0b: weights f32 -> f16
__global__ void k_cvt_w(const float* __restrict__ wq, const float* __restrict__ wp,
                        _Float16* __restrict__ wq16, _Float16* __restrict__ wp16) {
    int i = blockIdx.x * blockDim.x + threadIdx.x;
    if (i < 3 * C_SZ * C_SZ) wq16[i] = (_Float16)wq[i];
    if (i < C_SZ * C_SZ)     wp16[i] = (_Float16)wp[i];
}

// ---------------------------------------------------------------------------
// Kernel 1: QKV projection.  Out[o,l] = Wq[o,:] . x[:,l] + bias, o in [0,384).
// One wave per 16x16 (o,l) tile, 4 WMMAs over K=C=128.
// Scatter: q -> [bh,l,d]*scale, k -> [bh,l,d], v -> vT [bh,d,l]   (all f16)
// ---------------------------------------------------------------------------
#define QKV_OT (3 * C_SZ / 16)   // 24
#define QKV_LT (L_SEQ / 16)      // 128
__global__ void __launch_bounds__(256)
k_qkv(const _Float16* __restrict__ wq16, const _Float16* __restrict__ xT,
      const float* __restrict__ bqkv,
      _Float16* __restrict__ qh, _Float16* __restrict__ kh,
      _Float16* __restrict__ vT) {
    int wave = threadIdx.x >> 5;
    int tid  = blockIdx.x * (blockDim.x >> 5) + wave;
    int b    = tid / (QKV_OT * QKV_LT);
    int rem  = tid % (QKV_OT * QKV_LT);
    int ot   = rem / QKV_LT;
    int lt   = rem % QKV_LT;

    const _Float16* x = xT + (size_t)b * L_SEQ * C_SZ;
    v8f acc = {};
#pragma unroll
    for (int kc = 0; kc < C_SZ; kc += 32) {
        v16h a  = load_frag(wq16, C_SZ, ot * 16, kc);
        v16h bf = load_frag(x,    C_SZ, lt * 16, kc);
        acc = wmma16(a, bf, acc);
    }

    int lane = threadIdx.x & 31, khalf = lane >> 4, nl = lane & 15;
    int oBase = ot * 16;
    int which = oBase >> 7;           // 0=q 1=k 2=v (uniform per tile)
    int h     = (oBase & 127) >> 5;   // head (uniform per tile)
    int dbase = oBase & 31;           // 0 or 16 (uniform per tile)
    int bh    = b * H_NUM + h;
    int l     = lt * 16 + nl;

    int ob = oBase + 8 * khalf;
    float4 b0 = *(const float4*)(bqkv + ob);
    float4 b1 = *(const float4*)(bqkv + ob + 4);
    float bias[8] = {b0.x, b0.y, b0.z, b0.w, b1.x, b1.y, b1.z, b1.w};

#pragma unroll
    for (int j = 0; j < 8; ++j) {
        float val = acc[j] + bias[j];
        int d = dbase + j + 8 * khalf;
        if (which == 0)
            qh[((size_t)bh * L_SEQ + l) * D_HD + d] = (_Float16)(val * QSCALE);
        else if (which == 1)
            kh[((size_t)bh * L_SEQ + l) * D_HD + d] = (_Float16)val;
        else
            vT[((size_t)bh * D_HD + d) * L_SEQ + l] = (_Float16)val;
    }
}

// ---------------------------------------------------------------------------
// Kernel 2: flash attention, 64-key blocks.  One wave per 16 query rows.
//   S = Q.K^T : 4 WMMAs -> one softmax pass (row stats shuffled across the
//   16-lane half-wave) -> P through a wave-private 16x64 LDS tile (C/D-layout
//   -> A-layout transpose) -> P.V : 4 WMMAs against vT.
// Output Oh [B, L, C] f16  (c = h*32 + d, contiguous for the proj GEMM).
// ---------------------------------------------------------------------------
__global__ void __launch_bounds__(128)
k_attn(const _Float16* __restrict__ qh, const _Float16* __restrict__ kh,
       const _Float16* __restrict__ vT, _Float16* __restrict__ Oh) {
    __shared__ _Float16 pbuf[4][16 * 64];
    int wave  = threadIdx.x >> 5;
    int lane  = threadIdx.x & 31;
    int khalf = lane >> 4, nl = lane & 15;
    int bh = blockIdx.y;                 // 0..31
    int qt = blockIdx.x * 4 + wave;      // 0..127

    const _Float16* q = qh + (size_t)bh * L_SEQ * D_HD;
    const _Float16* k = kh + (size_t)bh * L_SEQ * D_HD;
    const _Float16* v = vT + (size_t)bh * D_HD * L_SEQ;

    v16h aq = load_frag(q, D_HD, qt * 16, 0);

    v8f o0 = {}, o1 = {};
    float mrow[8], lrow[8];
#pragma unroll
    for (int j = 0; j < 8; ++j) { mrow[j] = -1e30f; lrow[j] = 0.f; }

    _Float16* P = pbuf[wave];

    for (int kb = 0; kb < L_SEQ; kb += 64) {
        if (kb + 64 < L_SEQ) {
            __builtin_prefetch(k + (size_t)(kb + 64) * D_HD, 0, 3);
            __builtin_prefetch(v + (size_t)(kb + 64), 0, 3);
        }
        v16h bk0 = load_frag(k, D_HD, kb,      0);
        v16h bk1 = load_frag(k, D_HD, kb + 16, 0);
        v16h bk2 = load_frag(k, D_HD, kb + 32, 0);
        v16h bk3 = load_frag(k, D_HD, kb + 48, 0);
        v8f z = {};
        v8f s0 = wmma16(aq, bk0, z);
        v8f s1 = wmma16(aq, bk1, z);
        v8f s2 = wmma16(aq, bk2, z);
        v8f s3 = wmma16(aq, bk3, z);

#pragma unroll
        for (int j = 0; j < 8; ++j) {
            // row max across 64 keys: 3 local max + 4 half-wave shuffles
            float t = fmaxf(fmaxf(s0[j], s1[j]), fmaxf(s2[j], s3[j]));
            t = fmaxf(t, __shfl_xor(t, 1, 32));
            t = fmaxf(t, __shfl_xor(t, 2, 32));
            t = fmaxf(t, __shfl_xor(t, 4, 32));
            t = fmaxf(t, __shfl_xor(t, 8, 32));
            float mn   = fmaxf(mrow[j], t);
            float corr = __expf(mrow[j] - mn);
            mrow[j] = mn;
            float p0 = __expf(s0[j] - mn);
            float p1 = __expf(s1[j] - mn);
            float p2 = __expf(s2[j] - mn);
            float p3 = __expf(s3[j] - mn);
            float rs = (p0 + p1) + (p2 + p3);
            rs += __shfl_xor(rs, 1, 32);
            rs += __shfl_xor(rs, 2, 32);
            rs += __shfl_xor(rs, 4, 32);
            rs += __shfl_xor(rs, 8, 32);
            lrow[j] = lrow[j] * corr + rs;
            o0[j] *= corr;
            o1[j] *= corr;
            int row = j + 8 * khalf;
            P[row * 64 +      nl] = (_Float16)p0;
            P[row * 64 + 16 + nl] = (_Float16)p1;
            P[row * 64 + 32 + nl] = (_Float16)p2;
            P[row * 64 + 48 + nl] = (_Float16)p3;
        }
        asm volatile("s_wait_dscnt 0" ::: "memory");  // C/D->A transpose fence

        v16h pa0  = load_frag(P, 64, 0, 0);           // keys kb..kb+31
        v16h pa1  = load_frag(P, 64, 0, 32);          // keys kb+32..kb+63
        v16h bv00 = load_frag(v, L_SEQ, 0,  kb);      // d 0..15,  keys kb..+31
        v16h bv10 = load_frag(v, L_SEQ, 16, kb);      // d 16..31, keys kb..+31
        v16h bv01 = load_frag(v, L_SEQ, 0,  kb + 32); // d 0..15,  keys +32..63
        v16h bv11 = load_frag(v, L_SEQ, 16, kb + 32); // d 16..31, keys +32..63
        o0 = wmma16(pa0, bv00, o0);
        o1 = wmma16(pa0, bv10, o1);
        o0 = wmma16(pa1, bv01, o0);
        o1 = wmma16(pa1, bv11, o1);
    }

    int h = bh & (H_NUM - 1);
    int b = bh >> 2;
#pragma unroll
    for (int j = 0; j < 8; ++j) {
        int l = qt * 16 + j + 8 * khalf;
        float inv = 1.f / lrow[j];
        size_t base = ((size_t)(b * L_SEQ + l)) * C_SZ + h * D_HD;
        Oh[base + nl]      = (_Float16)(o0[j] * inv);
        Oh[base + 16 + nl] = (_Float16)(o1[j] * inv);
    }
}

// ---------------------------------------------------------------------------
// Kernel 3: output projection + bias + residual.  y = Wp.Oh + bp + x (f32 out)
// ---------------------------------------------------------------------------
#define PRJ_OT (C_SZ / 16)     // 8
#define PRJ_LT (L_SEQ / 16)    // 128
__global__ void __launch_bounds__(256)
k_proj(const _Float16* __restrict__ wp16, const _Float16* __restrict__ Oh,
       const float* __restrict__ bp, const float* __restrict__ x,
       float* __restrict__ out) {
    int wave = threadIdx.x >> 5;
    int tid  = blockIdx.x * (blockDim.x >> 5) + wave;
    int b    = tid / (PRJ_OT * PRJ_LT);
    int rem  = tid % (PRJ_OT * PRJ_LT);
    int ot   = rem / PRJ_LT;
    int lt   = rem % PRJ_LT;

    const _Float16* O = Oh + (size_t)b * L_SEQ * C_SZ;
    v8f acc = {};
#pragma unroll
    for (int kc = 0; kc < C_SZ; kc += 32) {
        v16h a  = load_frag(wp16, C_SZ, ot * 16, kc);
        v16h bf = load_frag(O,    C_SZ, lt * 16, kc);
        acc = wmma16(a, bf, acc);
    }

    int lane = threadIdx.x & 31, khalf = lane >> 4, nl = lane & 15;
    int l  = lt * 16 + nl;
    int ob = ot * 16 + 8 * khalf;
    float4 b0 = *(const float4*)(bp + ob);
    float4 b1 = *(const float4*)(bp + ob + 4);
    float bias[8] = {b0.x, b0.y, b0.z, b0.w, b1.x, b1.y, b1.z, b1.w};

#pragma unroll
    for (int j = 0; j < 8; ++j) {
        int o = ob + j;
        size_t idx = ((size_t)(b * C_SZ + o)) * L_SEQ + l;
        out[idx] = acc[j] + bias[j] + x[idx];
    }
}

// ---------------------------------------------------------------------------
// Host-side launch.  Workspace layout (bytes, 256-aligned), total ~20.2 MB:
//   wq16   @ 0         (96 KB)     wp16 @ 98304   (32 KB)
//   xT     @ 131072    (4 MB)      qh   @ 4325376 (4 MB)
//   kh     @ 8519680   (4 MB)      vT   @ 12713984(4 MB)
//   Oh     @ 16908288  (4 MB)
// ---------------------------------------------------------------------------
extern "C" void kernel_launch(void* const* d_in, const int* in_sizes, int n_in,
                              void* d_out, int out_size, void* d_ws, size_t ws_size,
                              hipStream_t stream) {
    const float* x     = (const float*)d_in[0];
    const float* w_qkv = (const float*)d_in[1];
    const float* b_qkv = (const float*)d_in[2];
    const float* w_prj = (const float*)d_in[3];
    const float* b_prj = (const float*)d_in[4];
    float* out = (float*)d_out;

    char* ws = (char*)d_ws;
    _Float16* wq16 = (_Float16*)(ws + 0);
    _Float16* wp16 = (_Float16*)(ws + 98304);
    _Float16* xT   = (_Float16*)(ws + 131072);
    _Float16* qh   = (_Float16*)(ws + 4325376);
    _Float16* kh   = (_Float16*)(ws + 8519680);
    _Float16* vT   = (_Float16*)(ws + 12713984);
    _Float16* Oh   = (_Float16*)(ws + 16908288);

    // 0) convert inputs to f16 (x also transposed to [B,L,C])
    {
        size_t n = (size_t)B_SZ * C_SZ * L_SEQ;   // 2,097,152
        k_cvt_x<<<dim3((unsigned)((n + 255) / 256)), dim3(256), 0, stream>>>(x, xT);
        k_cvt_w<<<dim3((3 * C_SZ * C_SZ + 255) / 256), dim3(256), 0, stream>>>(
            w_qkv, w_prj, wq16, wp16);
    }
    // 1) QKV projection: B * 24 * 128 = 24576 tiles, 8 waves/block
    k_qkv<<<dim3(B_SZ * QKV_OT * QKV_LT / 8), dim3(256), 0, stream>>>(
        wq16, xT, b_qkv, qh, kh, vT);
    // 2) attention: grid (L/64 q-groups, B*H)
    k_attn<<<dim3(L_SEQ / 64, B_SZ * H_NUM), dim3(128), 0, stream>>>(
        qh, kh, vT, Oh);
    // 3) projection + residual: B * 8 * 128 = 8192 tiles, 8 waves/block
    k_proj<<<dim3(B_SZ * PRJ_OT * PRJ_LT / 8), dim3(256), 0, stream>>>(
        wp16, Oh, b_prj, x, out);
}
